// GPT_Model_18674517803138
// MI455X (gfx1250) — matrix operand order repaired and checked by reading, not compiled, not run on hardware
//
#include <hip/hip_runtime.h>
#include <hip/hip_bf16.h>
#include <math.h>

// ---------------------------------------------------------------------------
// GPT forward pass for MI455X (gfx1250, wave32, WMMA).
// GEMMs + attention use v_wmma_f32_16x16x32_f16 (fp16 in, fp32 accumulate).
// Double-buffered LDS staging, register-batched global loads.
// ---------------------------------------------------------------------------

typedef __attribute__((ext_vector_type(16))) _Float16 v16h;
typedef __attribute__((ext_vector_type(8)))  float    v8f;

#define LNEG (-3.0e38f)

__device__ __forceinline__ v8f wmma16(v16h a, v16h b, v8f c) {
  // D = A(16x32 f16) * B(32x16 f16) + C(16x16 f32)
  return __builtin_amdgcn_wmma_f32_16x16x32_f16(
      /*neg_a=*/false, a, /*neg_b=*/false, b,
      /*c_mod=*/(short)0, c, /*reuse_a=*/false, /*reuse_b=*/false);
}

// A fragment: 16x32 tile, rows [row0,row0+16), K [k0,k0+32).
// LDS layout: row-major [rows][ldh] halves (ldh = 72 -> 144B row, 16B aligned).
__device__ __forceinline__ v16h load_a_frag(const _Float16* base, int row0,
                                            int k0, int ldh, int lane) {
  const _Float16* rp = base + (size_t)(row0 + (lane & 15)) * ldh + k0;
  const int off = (lane >> 4) << 4;
  v16h a;
#pragma unroll
  for (int j = 0; j < 8; ++j) {
    int k = 2 * j + off;
    a[2 * j]     = rp[k];
    a[2 * j + 1] = rp[k + 1];
  }
  return a;
}

// B fragment: 32x16 tile, cols [n0,n0+16), K [k0,k0+32).
// LDS layout: [n][k] with stride ldh halves.
__device__ __forceinline__ v16h load_b_frag(const _Float16* base, int n0,
                                            int k0, int ldh, int lane) {
  const _Float16* rp =
      base + (size_t)(n0 + (lane & 15)) * ldh + k0 + ((lane >> 4) << 4);
  v16h b;
#pragma unroll
  for (int e = 0; e < 16; ++e) b[e] = rp[e];
  return b;
}

__device__ __forceinline__ void st4h(_Float16* p, float4 v) {
  p[0] = (_Float16)v.x; p[1] = (_Float16)v.y;
  p[2] = (_Float16)v.z; p[3] = (_Float16)v.w;
}

// ---------------------------------------------------------------------------
// Embedding: X[b,t,:] = wte[idx[b,t],:] + wpe[t,:]
// ---------------------------------------------------------------------------
__global__ __launch_bounds__(256) void embed_kernel(
    const int* __restrict__ idx, const float* __restrict__ wte,
    const float* __restrict__ wpe, float* __restrict__ X) {
  const int row  = blockIdx.x;
  const int tok  = idx[row];
  const int tpos = row & 1023;
  const float* we = wte + (size_t)tok * 768;
  const float* pe = wpe + (size_t)tpos * 768;
  float* xr = X + (size_t)row * 768;
  for (int c = threadIdx.x; c < 768; c += 256) xr[c] = we[c] + pe[c];
}

// ---------------------------------------------------------------------------
// LayerNorm, one block per row of C elements.
// ---------------------------------------------------------------------------
__global__ __launch_bounds__(256) void ln_kernel(
    const float* __restrict__ x, const float* __restrict__ w,
    const float* __restrict__ b, float* __restrict__ out, int C) {
  __shared__ float red[256];
  const int row = blockIdx.x, t = threadIdx.x;
  const float* xr = x + (size_t)row * C;

  float s = 0.f;
  for (int c = t; c < C; c += 256) s += xr[c];
  red[t] = s; __syncthreads();
  for (int o = 128; o > 0; o >>= 1) { if (t < o) red[t] += red[t + o]; __syncthreads(); }
  const float mean = red[0] / (float)C;
  __syncthreads();

  float s2 = 0.f;
  for (int c = t; c < C; c += 256) { float d = xr[c] - mean; s2 += d * d; }
  red[t] = s2; __syncthreads();
  for (int o = 128; o > 0; o >>= 1) { if (t < o) red[t] += red[t + o]; __syncthreads(); }
  const float rstd = rsqrtf(red[0] / (float)C + 1e-5f);
  __syncthreads();

  float* orow = out + (size_t)row * C;
  for (int c = t; c < C; c += 256)
    orow[c] = (xr[c] - mean) * rstd * w[c] + b[c];
}

// ---------------------------------------------------------------------------
// WMMA GEMM: out[M,N] = act(A[M,K] * W + bias + residual)
//   wt==0: W stored [K,N] row-major; wt==1: W stored [N,K] row-major (x@W^T).
// 256 threads (8 waves), tile 64x64, K-step 64, double-buffered LDS.
// Wave w: rows (w&3)*16, col base (w>>2)*32 -> two 16x16 accumulators,
// 4 WMMA per wave per K-step.
// ---------------------------------------------------------------------------
__global__ __launch_bounds__(256) void gemm_kernel(
    const float* __restrict__ A, const float* __restrict__ W,
    const float* __restrict__ bias, const float* __restrict__ res,
    float* __restrict__ out, int M, int N, int K, int wt, int relu) {
  __shared__ _Float16 As[2][64][72];   // [buf][m][k]
  __shared__ _Float16 Bt[2][64][72];   // [buf][n][k]

  const int t    = threadIdx.x;
  const int lane = t & 31;
  const int w    = t >> 5;
  const int m0   = blockIdx.y * 64;
  const int n0   = blockIdx.x * 64;
  const int rm   = (w & 3) << 4;
  const int cn   = (w >> 2) << 5;

  // staging coords: 64x64 tile = 1024 float4, 4 per thread
  int arow[4], akq[4], br[4], bq[4];
  const float* ap[4];
  const float* bp[4];
#pragma unroll
  for (int i = 0; i < 4; ++i) {
    const int li = i * 256 + t;
    arow[i] = li >> 4;            // 0..63
    akq[i]  = (li & 15) << 2;     // 0,4,..,60
    ap[i] = A + (size_t)(m0 + arow[i]) * K + akq[i];
    br[i] = li >> 4;
    bq[i] = (li & 15) << 2;
    if (wt)  // W[N,K]: br = n row, bq = k quad (vector along k)
      bp[i] = W + (size_t)(n0 + br[i]) * K + bq[i];
    else     // W[K,N]: br = k row, bq = n quad (vector along n)
      bp[i] = W + (size_t)br[i] * N + n0 + bq[i];
  }
  const size_t bstep = wt ? (size_t)64 : (size_t)64 * N;

  float4 ra[4], rb[4];
  // ---- preload tile 0 (batch all global loads, then convert/store) ----
#pragma unroll
  for (int i = 0; i < 4; ++i) { ra[i] = *(const float4*)ap[i]; ap[i] += 64; }
#pragma unroll
  for (int i = 0; i < 4; ++i) { rb[i] = *(const float4*)bp[i]; bp[i] += bstep; }
#pragma unroll
  for (int i = 0; i < 4; ++i) st4h(&As[0][arow[i]][akq[i]], ra[i]);
  if (wt) {
#pragma unroll
    for (int i = 0; i < 4; ++i) st4h(&Bt[0][br[i]][bq[i]], rb[i]);
  } else {
#pragma unroll
    for (int i = 0; i < 4; ++i) {
      Bt[0][bq[i] + 0][br[i]] = (_Float16)rb[i].x;
      Bt[0][bq[i] + 1][br[i]] = (_Float16)rb[i].y;
      Bt[0][bq[i] + 2][br[i]] = (_Float16)rb[i].z;
      Bt[0][bq[i] + 3][br[i]] = (_Float16)rb[i].w;
    }
  }
  __syncthreads();

  v8f c0 = {}, c1 = {};
  const int ksteps = K >> 6;
  for (int kt = 0; kt < ksteps; ++kt) {
    const int  cur  = kt & 1;
    const bool more = (kt + 1 < ksteps);
    if (more) {  // issue next tile's loads before doing the math
#pragma unroll
      for (int i = 0; i < 4; ++i) { ra[i] = *(const float4*)ap[i]; ap[i] += 64; }
#pragma unroll
      for (int i = 0; i < 4; ++i) { rb[i] = *(const float4*)bp[i]; bp[i] += bstep; }
    }
#pragma unroll
    for (int s = 0; s < 2; ++s) {
      v16h a  = load_a_frag(&As[cur][0][0], rm, s * 32, 72, lane);
      v16h b0 = load_b_frag(&Bt[cur][0][0], cn, s * 32, 72, lane);
      v16h b1 = load_b_frag(&Bt[cur][0][0], cn + 16, s * 32, 72, lane);
      c0 = wmma16(a, b0, c0);
      c1 = wmma16(a, b1, c1);
    }
    if (more) {
      const int nxt = cur ^ 1;
#pragma unroll
      for (int i = 0; i < 4; ++i) st4h(&As[nxt][arow[i]][akq[i]], ra[i]);
      if (wt) {
#pragma unroll
        for (int i = 0; i < 4; ++i) st4h(&Bt[nxt][br[i]][bq[i]], rb[i]);
      } else {
#pragma unroll
        for (int i = 0; i < 4; ++i) {
          Bt[nxt][bq[i] + 0][br[i]] = (_Float16)rb[i].x;
          Bt[nxt][bq[i] + 1][br[i]] = (_Float16)rb[i].y;
          Bt[nxt][bq[i] + 2][br[i]] = (_Float16)rb[i].z;
          Bt[nxt][bq[i] + 3][br[i]] = (_Float16)rb[i].w;
        }
      }
    }
    __syncthreads();
  }

  // epilogue: C layout lane->N = lane&15, VGPR r -> M = r + 8*(lane>=16)
#pragma unroll
  for (int f = 0; f < 2; ++f) {
    const int gn = n0 + cn + f * 16 + (lane & 15);
    const float bb = bias ? bias[gn] : 0.f;
#pragma unroll
    for (int r = 0; r < 8; ++r) {
      const int gm = m0 + rm + r + ((lane >> 4) << 3);
      float v = (f == 0 ? c0[r] : c1[r]) + bb;
      if (res) v += res[(size_t)gm * N + gn];
      if (relu) v = fmaxf(v, 0.f);
      out[(size_t)gm * N + gn] = v;
    }
  }
}

// ---------------------------------------------------------------------------
// Flash attention (causal), HS=64, H=12, T=1024, C=768.
// grid = (T/64, B*H). One block = one (b,h) + 64 q-rows, 8 waves.
// ---------------------------------------------------------------------------
__global__ __launch_bounds__(256) void attn_kernel(
    const float* __restrict__ Q, const float* __restrict__ K,
    const float* __restrict__ V, float* __restrict__ O) {
  __shared__ _Float16 Qs[64][72];   // [q][d], pre-scaled by 1/8
  __shared__ _Float16 Ks[64][72];   // [key][d]  (B-frag layout for Q.K^T)
  __shared__ _Float16 Vs[64][72];   // [d][key]  (B-frag layout for P.V)
  __shared__ _Float16 Ps[64][72];   // [q][key]  (A-frag source)
  __shared__ float    Sb[64][66];
  __shared__ float    pbuf[64][4];
  __shared__ float    mrow[64], lrow[64], arow[64];

  const int qt = blockIdx.x;
  const int bh = blockIdx.y;
  const int b  = bh / 12, h = bh % 12;
  const int C  = 768;
  const int t = threadIdx.x, lane = t & 31, w = t >> 5;
  const int rm = (w & 3) << 4;
  const int cn = (w >> 2) << 5;

  // ---- stage Q (batched loads, then convert) ----
  const float* Qb = Q + ((size_t)b * 1024 + qt * 64) * C + h * 64;
  {
    float4 rq[4];
#pragma unroll
    for (int i = 0; i < 4; ++i) {
      int li = i * 256 + t;
      rq[i] = *(const float4*)(Qb + (size_t)(li >> 4) * C + ((li & 15) << 2));
    }
#pragma unroll
    for (int i = 0; i < 4; ++i) {
      int li = i * 256 + t;
      float4 v = rq[i];
      v.x *= 0.125f; v.y *= 0.125f; v.z *= 0.125f; v.w *= 0.125f;
      st4h(&Qs[li >> 4][(li & 15) << 2], v);
    }
  }
  if (t < 64) { mrow[t] = LNEG; lrow[t] = 0.f; }

  v8f o0 = {}, o1 = {};
  const float* Kb = K + ((size_t)b * 1024) * C + h * 64;
  const float* Vb = V + ((size_t)b * 1024) * C + h * 64;
  __syncthreads();

  for (int kc = 0; kc <= qt; ++kc) {
    // ---- stage K (natural) and V (transposed) for this key chunk ----
    {
      float4 rk[4], rv[4];
#pragma unroll
      for (int i = 0; i < 4; ++i) {
        int li = i * 256 + t;
        const size_t go = (size_t)(li >> 4) * C + ((li & 15) << 2);
        rk[i] = *(const float4*)(Kb + go);
        rv[i] = *(const float4*)(Vb + go);
      }
#pragma unroll
      for (int i = 0; i < 4; ++i) {
        int li = i * 256 + t;
        int row = li >> 4, dq = (li & 15) << 2;
        st4h(&Ks[row][dq], rk[i]);
        Vs[dq + 0][row] = (_Float16)rv[i].x;
        Vs[dq + 1][row] = (_Float16)rv[i].y;
        Vs[dq + 2][row] = (_Float16)rv[i].z;
        Vs[dq + 3][row] = (_Float16)rv[i].w;
      }
    }
    Kb += (size_t)64 * C;
    Vb += (size_t)64 * C;
    __syncthreads();

    // ---- S(16x32 per wave) = Q . K^T over d=64 (two K-steps) ----
    v8f s0 = {}, s1 = {};
#pragma unroll
    for (int dk = 0; dk < 64; dk += 32) {
      v16h a  = load_a_frag(&Qs[0][0], rm, dk, 72, lane);
      v16h b0 = load_b_frag(&Ks[0][0], cn, dk, 72, lane);
      v16h b1 = load_b_frag(&Ks[0][0], cn + 16, dk, 72, lane);
      s0 = wmma16(a, b0, s0);
      s1 = wmma16(a, b1, s1);
    }
    // write S with causal mask
#pragma unroll
    for (int f = 0; f < 2; ++f) {
      const int kl = cn + f * 16 + (lane & 15);
      const int kg = kc * 64 + kl;
#pragma unroll
      for (int r = 0; r < 8; ++r) {
        const int ql = rm + r + ((lane >> 4) << 3);
        const int qg = qt * 64 + ql;
        float sv = (f == 0 ? s0[r] : s1[r]);
        Sb[ql][kl] = (kg > qg) ? LNEG : sv;
      }
    }
    __syncthreads();

    // ---- online softmax: 4 threads per q-row ----
    {
      const int row = t >> 2, sub = t & 3;
      const int e0 = sub << 4;
      float pm = LNEG;
#pragma unroll
      for (int e = 0; e < 16; ++e) pm = fmaxf(pm, Sb[row][e0 + e]);
      pbuf[row][sub] = pm;
      __syncthreads();
      const float mn = fmaxf(mrow[row],
                             fmaxf(fmaxf(pbuf[row][0], pbuf[row][1]),
                                   fmaxf(pbuf[row][2], pbuf[row][3])));
      float rs = 0.f;
#pragma unroll
      for (int e = 0; e < 16; ++e) {
        float p = __expf(Sb[row][e0 + e] - mn);
        Ps[row][e0 + e] = (_Float16)p;
        rs += p;
      }
      pbuf[row][sub] = rs;
      __syncthreads();
      if (sub == 0) {
        const float tot = pbuf[row][0] + pbuf[row][1] + pbuf[row][2] + pbuf[row][3];
        const float al = __expf(mrow[row] - mn);
        lrow[row] = lrow[row] * al + tot;
        mrow[row] = mn;
        arow[row] = al;
      }
    }
    __syncthreads();

    // ---- rescale O, then O += P.V (K dim = 64 keys, two K-steps) ----
#pragma unroll
    for (int r = 0; r < 8; ++r) {
      const float al = arow[rm + r + ((lane >> 4) << 3)];
      o0[r] *= al;
      o1[r] *= al;
    }
#pragma unroll
    for (int kk = 0; kk < 64; kk += 32) {
      v16h a  = load_a_frag(&Ps[0][0], rm, kk, 72, lane);
      v16h b0 = load_b_frag(&Vs[0][0], cn, kk, 72, lane);
      v16h b1 = load_b_frag(&Vs[0][0], cn + 16, kk, 72, lane);
      o0 = wmma16(a, b0, o0);
      o1 = wmma16(a, b1, o1);
    }
    __syncthreads();
  }

  // final normalize + store
  float* Ob = O + ((size_t)b * 1024 + qt * 64) * C + h * 64;
#pragma unroll
  for (int f = 0; f < 2; ++f) {
    const int col = cn + f * 16 + (lane & 15);
#pragma unroll
    for (int r = 0; r < 8; ++r) {
      const int ql = rm + r + ((lane >> 4) << 3);
      const float inv = 1.f / lrow[ql];
      Ob[(size_t)ql * C + col] = (f == 0 ? o0[r] : o1[r]) * inv;
    }
  }
}

// ---------------------------------------------------------------------------
// Per-row NLL: nll[row] = logsumexp(logits[row,:]) - logits[row, target]
// ---------------------------------------------------------------------------
__global__ __launch_bounds__(256) void nll_kernel(
    const float* __restrict__ logits, const int* __restrict__ targets,
    float* __restrict__ nll) {
  __shared__ float red[256];
  const int row = blockIdx.x, t = threadIdx.x;
  const float* lr = logits + (size_t)row * 1024;

  float mx = LNEG;
  for (int c = t; c < 1024; c += 256) mx = fmaxf(mx, lr[c]);
  red[t] = mx; __syncthreads();
  for (int o = 128; o > 0; o >>= 1) { if (t < o) red[t] = fmaxf(red[t], red[t + o]); __syncthreads(); }
  const float M = red[0];
  __syncthreads();

  float s = 0.f;
  for (int c = t; c < 1024; c += 256) s += __expf(lr[c] - M);
  red[t] = s; __syncthreads();
  for (int o = 128; o > 0; o >>= 1) { if (t < o) red[t] += red[t + o]; __syncthreads(); }

  if (t == 0) nll[row] = (__logf(red[0]) + M) - lr[targets[row]];
}

__global__ __launch_bounds__(256) void loss_kernel(
    const float* __restrict__ nll, float* __restrict__ out) {
  __shared__ float red[256];
  const int t = threadIdx.x;
  float s = 0.f;
  for (int i = t; i < 4096; i += 256) s += nll[i];
  red[t] = s; __syncthreads();
  for (int o = 128; o > 0; o >>= 1) { if (t < o) red[t] += red[t + o]; __syncthreads(); }
  if (t == 0) out[0] = red[0] / 4096.f;
}

// ---------------------------------------------------------------------------
// Host orchestration
// ---------------------------------------------------------------------------
extern "C" void kernel_launch(void* const* d_in, const int* in_sizes, int n_in,
                              void* d_out, int out_size, void* d_ws,
                              size_t ws_size, hipStream_t stream) {
  const int*   idx     = (const int*)d_in[0];
  const int*   targets = (const int*)d_in[1];
  const float* wte     = (const float*)d_in[2];
  const float* wpe     = (const float*)d_in[3];
  const float* lm_b    = (const float*)d_in[4];
  const float* ln1_w   = (const float*)d_in[5];
  const float* ln1_b   = (const float*)d_in[6];
  const float* Wq      = (const float*)d_in[7];
  const float* Wk      = (const float*)d_in[8];
  const float* Wv      = (const float*)d_in[9];
  const float* proj_W  = (const float*)d_in[10];
  const float* proj_b  = (const float*)d_in[11];
  const float* ln2_w   = (const float*)d_in[12];
  const float* ln2_b   = (const float*)d_in[13];
  const float* fc1_W   = (const float*)d_in[14];
  const float* fc1_b   = (const float*)d_in[15];
  const float* fc2_W   = (const float*)d_in[16];
  const float* fc2_b   = (const float*)d_in[17];
  const float* lnf_w   = (const float*)d_in[18];
  const float* lnf_b   = (const float*)d_in[19];

  const int Mr = 4096, C = 768, F = 3072, Vv = 1024, L = 6, H = 12, T = 1024;

  float* ws = (float*)d_ws;
  size_t o = 0;
  float* X    = ws + o; o += (size_t)Mr * C;
  float* Hb   = ws + o; o += (size_t)Mr * C;
  float* Qb   = ws + o; o += (size_t)Mr * C;
  float* Kb   = ws + o; o += (size_t)Mr * C;
  float* Vb   = ws + o; o += (size_t)Mr * C;
  float* ATT  = ws + o; o += (size_t)Mr * C;
  float* FFN  = ws + o; o += (size_t)Mr * F;
  float* NLL  = ws + o; o += (size_t)Mr;

  float* logits = (float*)d_out;                 // [4096,1024]
  float* loss   = logits + (size_t)Mr * Vv;      // scalar at tail

  const dim3 blk(256);

  embed_kernel<<<Mr, blk, 0, stream>>>(idx, wte, wpe, X);

  for (int l = 0; l < L; ++l) {
    ln_kernel<<<Mr, blk, 0, stream>>>(X, ln1_w + (size_t)l * C,
                                      ln1_b + (size_t)l * C, Hb, C);
    const dim3 gC(C / 64, Mr / 64);
    gemm_kernel<<<gC, blk, 0, stream>>>(Hb, Wq + (size_t)l * C * C, nullptr,
                                        nullptr, Qb, Mr, C, C, 0, 0);
    gemm_kernel<<<gC, blk, 0, stream>>>(Hb, Wk + (size_t)l * C * C, nullptr,
                                        nullptr, Kb, Mr, C, C, 0, 0);
    gemm_kernel<<<gC, blk, 0, stream>>>(Hb, Wv + (size_t)l * C * C, nullptr,
                                        nullptr, Vb, Mr, C, C, 0, 0);
    attn_kernel<<<dim3(T / 64, 4 * H), blk, 0, stream>>>(Qb, Kb, Vb, ATT);
    gemm_kernel<<<gC, blk, 0, stream>>>(ATT, proj_W + (size_t)l * C * C,
                                        proj_b + (size_t)l * C, X, X, Mr, C, C,
                                        0, 0);
    ln_kernel<<<Mr, blk, 0, stream>>>(X, ln2_w + (size_t)l * C,
                                      ln2_b + (size_t)l * C, Hb, C);
    gemm_kernel<<<dim3(F / 64, Mr / 64), blk, 0, stream>>>(
        Hb, fc1_W + (size_t)l * C * F, fc1_b + (size_t)l * F, nullptr, FFN, Mr,
        F, C, 0, 1);
    gemm_kernel<<<gC, blk, 0, stream>>>(FFN, fc2_W + (size_t)l * F * C,
                                        fc2_b + (size_t)l * C, X, X, Mr, C, F,
                                        0, 0);
  }

  ln_kernel<<<Mr, blk, 0, stream>>>(X, lnf_w, lnf_b, Hb, C);
  gemm_kernel<<<dim3(Vv / 64, Mr / 64), blk, 0, stream>>>(
      Hb, wte, lm_b, nullptr, logits, Mr, Vv, C, /*wt=*/1, 0);

  nll_kernel<<<Mr, blk, 0, stream>>>(logits, targets, NLL);
  loss_kernel<<<1, blk, 0, stream>>>(NLL, loss);
}
